// VDNet_15951508537396
// MI455X (gfx1250) — matrix operand order repaired and checked by reading, not compiled, not run on hardware
//
#include <hip/hip_runtime.h>
#include <math.h>

// ---------------------------------------------------------------------------
// CDNA5 (gfx1250) wave32 WMMA types
// ---------------------------------------------------------------------------
typedef __attribute__((ext_vector_type(16))) __bf16 v16bf;
typedef __attribute__((ext_vector_type(8)))  float  v8f;

__device__ __forceinline__ __bf16 f2bf(float f) { return (__bf16)f; }

// ---------------------------------------------------------------------------
// Generic same-pad KSxKS conv as implicit GEMM on v_wmma_f32_16x16x32_bf16.
//   GEMM: M = Cout, N = 16*OH*OW (batch fixed 16), K = Cin*KS*KS
//   Register blocking: one wave computes a (16*MT) x 16 tile of D, with MT
//   accumulators sharing a single B fragment per k-step (amortizes the
//   scattered im2col gather MT times).
//   Fused maxpool: pool^2 GEMM passes, elementwise max of f32 accumulators.
// Fragment layouts per CDNA5 ISA 7.12.2:
//   A (16x32 bf16): lane L -> row M=L&15; half h -> K = (h<8?h:h+8) + 8*(L>=16)
//   B (32x16 bf16): lane L -> col N=L&15; half h -> K = (h<8?h:h+8) + 8*(L>=16)
//   C/D (16x16 f32): lane L -> N=L&15; vgpr r -> M = r + 8*(L>=16)
// ---------------------------------------------------------------------------
template<int KS, int MT>
__global__ __launch_bounds__(32)
void conv_wmma_k(const float* __restrict__ x, const float* __restrict__ w,
                 const float* __restrict__ bias, float* __restrict__ y,
                 int Cin, int H, int W, int Cout, int pool, int relu)
{
  constexpr int KK = KS * KS;
  const int OH  = H / pool, OW = W / pool;
  const int K   = Cin * KK;
  const int pad = (KS - 1) / 2;

  const int lane = threadIdx.x & 31;
  const int hi   = lane >> 4;
  const int lm   = lane & 15;

  // this lane's GEMM column (output pixel) for B and D fragments
  const int n   = blockIdx.x * 16 + lm;
  const int ohw = OH * OW;
  const int bb  = n / ohw;
  const int rem = n - bb * ohw;
  const int oy  = rem / OW;
  const int ox  = rem - oy * OW;

  const int  co_base = blockIdx.y * (16 * MT);
  const bool vecA    = ((K & 31) == 0);   // contiguous, aligned A fast path

  v8f best[MT];
#pragma unroll
  for (int mt = 0; mt < MT; ++mt)
#pragma unroll
    for (int i = 0; i < 8; ++i) best[mt][i] = -3.0e38f;

  const int npass = pool * pool;
  for (int p = 0; p < npass; ++p) {
    const int py = p / pool;
    const int cy = oy * pool + py;              // full-res conv output coords
    const int cx = ox * pool + (p - py * pool);

    v8f acc[MT];
#pragma unroll
    for (int mt = 0; mt < MT; ++mt)
#pragma unroll
      for (int i = 0; i < 8; ++i) acc[mt][i] = 0.f;

    for (int k0 = 0; k0 < K; k0 += 32) {
      // ---- B fragment (shared by all MT accumulators): im2col gather ----
      v16bf bfr;
      const int kb = k0 + hi * 8;
#pragma unroll
      for (int h = 0; h < 16; ++h) {
        int k = kb + h + (h >= 8 ? 8 : 0);
        float v = 0.f;
        if (k < K) {
          int ci = k / KK;
          int r2 = k - ci * KK;
          int ky = r2 / KS;
          int kx = r2 - ky * KS;
          int iy = cy + ky - pad;
          int ix = cx + kx - pad;
          if ((unsigned)iy < (unsigned)H && (unsigned)ix < (unsigned)W)
            v = x[(((size_t)bb * Cin + ci) * H + iy) * W + ix];
        }
        bfr[h] = f2bf(v);
      }

      // ---- MT stacked A fragments / WMMAs ----
#pragma unroll
      for (int mt = 0; mt < MT; ++mt) {
        const int co0m = co_base + mt * 16;
        if (co0m >= Cout) continue;             // wave-uniform scalar skip
        const int  coA  = co0m + lm;
        const bool aOk  = coA < Cout;
        const float* wrow = w + (size_t)coA * (size_t)K;

        v16bf a;
        if (vecA && aOk) {
          const float* w0 = wrow + k0 + hi * 8;      // h=0..7
          const float* w1 = wrow + k0 + 16 + hi * 8; // h=8..15
#pragma unroll
          for (int h = 0; h < 8; ++h) a[h] = f2bf(w0[h]);
#pragma unroll
          for (int h = 0; h < 8; ++h) a[h + 8] = f2bf(w1[h]);
        } else {
#pragma unroll
          for (int h = 0; h < 16; ++h) {
            int k = (h < 8) ? (k0 + hi * 8 + h) : (k0 + 16 + hi * 8 + (h - 8));
            float v = (aOk && k < K) ? wrow[k] : 0.f;
            a[h] = f2bf(v);
          }
        }

        acc[mt] = __builtin_amdgcn_wmma_f32_16x16x32_bf16(
            /*neg_a=*/false, a, /*neg_b=*/false, bfr,
            /*c_mod=*/(short)0, acc[mt], /*reuse_a=*/false, /*reuse_b=*/false);
      }
    }

#pragma unroll
    for (int mt = 0; mt < MT; ++mt)
#pragma unroll
      for (int i = 0; i < 8; ++i) best[mt][i] = fmaxf(best[mt][i], acc[mt][i]);
  }

  // ---- epilogue: bias (+relu) and store D ----
#pragma unroll
  for (int mt = 0; mt < MT; ++mt) {
    const int co0m = co_base + mt * 16;
    if (co0m >= Cout) continue;
#pragma unroll
    for (int r = 0; r < 8; ++r) {
      int co = co0m + r + 8 * hi;
      if (co < Cout) {
        float res = best[mt][r] + bias[co];
        if (relu) res = fmaxf(res, 0.f);
        y[(((size_t)bb * Cout + co) * OH + oy) * OW + ox] = res;
      }
    }
  }
}

// ---------------------------------------------------------------------------
// Small supporting kernels (VALU work, negligible FLOPs)
// ---------------------------------------------------------------------------

// concat two [16,3,128,128] frames into [16,6,128,128]
__global__ void concat_frames_k(const float* __restrict__ a,
                                const float* __restrict__ b,
                                float* __restrict__ o) {
  int i = blockIdx.x * blockDim.x + threadIdx.x;
  if (i >= 16 * 6 * 128 * 128) return;
  int pix = i & 16383;
  int c   = (i >> 14) % 6;
  int nn  = i / (6 * 16384);
  o[i] = (c < 3) ? a[(nn * 3 + c) * 16384 + pix]
                 : b[(nn * 3 + (c - 3)) * 16384 + pix];
}

// z = mean + exp(0.5*log_var)*eps   (m: [16,256] -> mean|log_var)
__global__ void vae_z_k(const float* __restrict__ m,
                        const float* __restrict__ eps,
                        float* __restrict__ z) {
  int i = blockIdx.x * blockDim.x + threadIdx.x;
  if (i >= 2048) return;
  int b = i >> 7, j = i & 127;
  float mean = m[b * 256 + j];
  float lv   = m[b * 256 + 128 + j];
  z[i] = mean + expf(0.5f * lv) * eps[i];
}

// h[b,o,y,x] = sum_i z[b,i]*Wd[i,o,y,x] + bd[o]   (Wd: [128,128,5,5])
__global__ void kdec_deconv_k(const float* __restrict__ z,
                              const float* __restrict__ Wd,
                              const float* __restrict__ bd,
                              float* __restrict__ h) {
  int i = blockIdx.x * blockDim.x + threadIdx.x;
  if (i >= 16 * 128 * 25) return;
  int s = i % 25, o = (i / 25) % 128, b = i / 3200;
  float acc = bd[o];
  const float* zp = z + b * 128;
  for (int k = 0; k < 128; ++k)
    acc += zp[k] * Wd[(k * 128 + o) * 25 + s];
  h[i] = acc;
}

// per-channel mean/var over (B=16, 5, 5) = 400 elems; one wave per channel
__global__ void bn_stats_k(const float* __restrict__ h, float* __restrict__ muvar) {
  int c = blockIdx.x;           // 0..127
  int lane = threadIdx.x & 31;
  float s = 0.f, s2 = 0.f;
  for (int t = lane; t < 400; t += 32) {
    int b = t / 25, e = t - (t / 25) * 25;
    float v = h[(b * 128 + c) * 25 + e];
    s += v; s2 += v * v;
  }
  for (int off = 16; off > 0; off >>= 1) {
    s  += __shfl_xor(s,  off, 32);
    s2 += __shfl_xor(s2, off, 32);
  }
  if (lane == 0) {
    float mu = s * (1.f / 400.f);
    muvar[c]       = mu;
    muvar[128 + c] = s2 * (1.f / 400.f) - mu * mu;
  }
}

// kernels = g*(h-mu)*rsqrt(var+1e-5)+be  (layout [16,128,5,5] == [16,4,32,1,5,5])
__global__ void bn_apply_k(const float* __restrict__ h,
                           const float* __restrict__ muvar,
                           const float* __restrict__ g,
                           const float* __restrict__ be,
                           float* __restrict__ kern) {
  int i = blockIdx.x * blockDim.x + threadIdx.x;
  if (i >= 51200) return;
  int c = (i / 25) % 128;
  float mu = muvar[c], var = muvar[128 + c];
  kern[i] = g[c] * (h[i] - mu) * rsqrtf(var + 1e-5f) + be[c];
}

// per-sample depthwise 5x5 cross-correlation, pad 2.
// f: [16,32,S,S]; kern: [16,4,32,5,5] (use scale slice)
__global__ void dyn_conv_k(const float* __restrict__ f,
                           const float* __restrict__ kern,
                           float* __restrict__ o, int S, int scale) {
  int i = blockIdx.x * blockDim.x + threadIdx.x;
  int total = 16 * 32 * S * S;
  if (i >= total) return;
  int x = i % S, y = (i / S) % S, c = (i / (S * S)) % 32, nn = i / (S * S * 32);
  const float* kp = kern + ((nn * 4 + scale) * 32 + c) * 25;
  const float* fp = f + (size_t)(nn * 32 + c) * S * S;
  float acc = 0.f;
#pragma unroll
  for (int ky = 0; ky < 5; ++ky) {
    int iy = y + ky - 2;
    if ((unsigned)iy >= (unsigned)S) continue;
#pragma unroll
    for (int kx = 0; kx < 5; ++kx) {
      int ix = x + kx - 2;
      if ((unsigned)ix < (unsigned)S) acc += fp[iy * S + ix] * kp[ky * 5 + kx];
    }
  }
  o[i] = acc;
}

// nearest-upsample each feat to 64x64 and concat along channels -> [16,128,64,64]
__global__ void upcat_k(const float* __restrict__ f1, const float* __restrict__ f2,
                        const float* __restrict__ f3, const float* __restrict__ f4,
                        float* __restrict__ o) {
  int i = blockIdx.x * blockDim.x + threadIdx.x;
  if (i >= 16 * 128 * 64 * 64) return;
  int x = i & 63, y = (i >> 6) & 63, c = (i >> 12) & 127, nn = i >> 19;
  int sidx = c >> 5, cc = c & 31;
  const float* fp; int S, sh;
  if (sidx == 0)      { fp = f1; S = 8;  sh = 3; }
  else if (sidx == 1) { fp = f2; S = 16; sh = 2; }
  else if (sidx == 2) { fp = f3; S = 32; sh = 1; }
  else                { fp = f4; S = 64; sh = 0; }
  o[i] = fp[((nn * 32 + cc) * S + (y >> sh)) * S + (x >> sh)];
}

// ---------------------------------------------------------------------------
// Orchestration
// ---------------------------------------------------------------------------
extern "C" void kernel_launch(void* const* d_in, const int* in_sizes, int n_in,
                              void* d_out, int out_size, void* d_ws, size_t ws_size,
                              hipStream_t stream)
{
  (void)in_sizes; (void)n_in; (void)out_size; (void)ws_size;

  const float* img[4]  = {(const float*)d_in[0], (const float*)d_in[1],
                          (const float*)d_in[2], (const float*)d_in[3]};
  const float* frame_a = (const float*)d_in[4];
  const float* frame_b = (const float*)d_in[5];
  const float* eps     = (const float*)d_in[6];

  // params flattened as jax pytree (dict keys sorted):
  // img_enc(32), kdec_bn(2), kdec_conv(4), kdec_deconv(2), mdec(6), mot_enc(12)
  int pi = 7;
  const float *ie_w[4][4], *ie_b[4][4];
  for (int s = 0; s < 4; ++s)
    for (int l = 0; l < 4; ++l) {
      ie_w[s][l] = (const float*)d_in[pi++];
      ie_b[s][l] = (const float*)d_in[pi++];
    }
  const float* bn_g  = (const float*)d_in[pi++];
  const float* bn_be = (const float*)d_in[pi++];
  const float *kc_w[2], *kc_b[2];
  for (int l = 0; l < 2; ++l) { kc_w[l] = (const float*)d_in[pi++]; kc_b[l] = (const float*)d_in[pi++]; }
  const float* kd_w = (const float*)d_in[pi++];
  const float* kd_b = (const float*)d_in[pi++];
  const float *md_w[3], *md_b[3];
  for (int l = 0; l < 3; ++l) { md_w[l] = (const float*)d_in[pi++]; md_b[l] = (const float*)d_in[pi++]; }
  const float *me_w[6], *me_b[6];
  for (int l = 0; l < 6; ++l) { me_w[l] = (const float*)d_in[pi++]; me_b[l] = (const float*)d_in[pi++]; }

  // workspace layout (floats); big buffers ping-pong across stages
  float* ws = (float*)d_ws;
  size_t off = 0;
  auto alloc = [&](size_t nf) { float* p = ws + off; off += nf; return p; };
  float* BUF0 = alloc(16777216);   // 64 MiB
  float* BUF1 = alloc(16777216);   // 64 MiB
  const int FS[4] = {8, 16, 32, 64};
  float* FIN[4];  for (int s = 0; s < 4; ++s) FIN[s]  = alloc((size_t)16 * 32 * FS[s] * FS[s]);
  float* FOUT[4]; for (int s = 0; s < 4; ++s) FOUT[s] = alloc((size_t)16 * 32 * FS[s] * FS[s]);
  float* M6   = alloc(16 * 256);
  float* Z    = alloc(16 * 128);
  float* H0   = alloc(51200);
  float* H1   = alloc(51200);
  float* H2   = alloc(51200);
  float* MUV  = alloc(256);
  float* KERN = alloc(51200);

  auto conv5 = [&](const float* x, const float* w, const float* b, float* y,
                   int Cin, int H, int Cout, int pool, int relu) {
    int OH = H / pool;
    dim3 g((16 * OH * OH) / 16, (Cout + 63) / 64);
    conv_wmma_k<5, 4><<<g, dim3(32), 0, stream>>>(x, w, b, y, Cin, H, H, Cout, pool, relu);
  };

  // ---- image encoder pyramid -> FIN[s] = [16,32,S/4,S/4] ----
  for (int s = 0; s < 4; ++s) {
    int S = 32 << s;
    conv5(img[s], ie_w[s][0], ie_b[s][0], BUF0,   3, S,     64, 2, 1);
    conv5(BUF0,   ie_w[s][1], ie_b[s][1], BUF1,  64, S / 2, 64, 1, 1);
    conv5(BUF1,   ie_w[s][2], ie_b[s][2], BUF0,  64, S / 2, 64, 2, 1);
    conv5(BUF0,   ie_w[s][3], ie_b[s][3], FIN[s], 64, S / 4, 32, 1, 1);
  }

  // ---- motion encoder (VAE) ----
  concat_frames_k<<<(16 * 6 * 128 * 128 + 255) / 256, 256, 0, stream>>>(frame_a, frame_b, BUF0);
  conv5(BUF0, me_w[0], me_b[0], BUF1,   6, 128,  96, 4, 1);
  conv5(BUF1, me_w[1], me_b[1], BUF0,  96,  32,  96, 2, 1);
  conv5(BUF0, me_w[2], me_b[2], BUF1,  96,  16, 128, 2, 1);
  conv5(BUF1, me_w[3], me_b[3], BUF0, 128,   8, 128, 2, 1);
  conv5(BUF0, me_w[4], me_b[4], BUF1, 128,   4, 256, 2, 1);
  conv5(BUF1, me_w[5], me_b[5], M6,   256,   2, 256, 2, 0);

  vae_z_k<<<8, 256, 0, stream>>>(M6, eps, Z);

  // ---- kernel decoder -> per-sample depthwise kernels ----
  kdec_deconv_k<<<200, 256, 0, stream>>>(Z, kd_w, kd_b, H0);
  conv5(H0, kc_w[0], kc_b[0], H1, 128, 5, 128, 1, 1);
  conv5(H1, kc_w[1], kc_b[1], H2, 128, 5, 128, 1, 0);
  bn_stats_k<<<128, 32, 0, stream>>>(H2, MUV);
  bn_apply_k<<<200, 256, 0, stream>>>(H2, MUV, bn_g, bn_be, KERN);

  // ---- dynamic per-sample depthwise conv on each scale ----
  for (int s = 0; s < 4; ++s) {
    int total = 16 * 32 * FS[s] * FS[s];
    dyn_conv_k<<<(total + 255) / 256, 256, 0, stream>>>(FIN[s], KERN, FOUT[s], FS[s], s);
  }

  // ---- motion decoder ----
  float* CAT = BUF0;
  float* MD1 = BUF1;
  float* MD2 = BUF0 + 8388608;
  upcat_k<<<(16 * 128 * 64 * 64 + 255) / 256, 256, 0, stream>>>(FOUT[0], FOUT[1], FOUT[2], FOUT[3], CAT);
  {
    dim3 g((16 * 64 * 64) / 16, 2);
    conv_wmma_k<9, 4><<<g, dim3(32), 0, stream>>>(CAT, md_w[0], md_b[0], MD1, 128, 64, 64, 128, 1, 1);
  }
  {
    dim3 g((16 * 64 * 64) / 16, 2);
    conv_wmma_k<1, 4><<<g, dim3(32), 0, stream>>>(MD1, md_w[1], md_b[1], MD2, 128, 64, 64, 128, 1, 1);
  }
  {
    dim3 g((16 * 64 * 64) / 16, 1);
    conv_wmma_k<1, 4><<<g, dim3(32), 0, stream>>>(MD2, md_w[2], md_b[2], (float*)d_out, 128, 64, 64, 3, 1, 0);
  }
}
